// VoxelBackBone8x_63677185130615
// MI455X (gfx1250) — matrix operand order repaired and checked by reading, not compiled, not run on hardware
//
#include <hip/hip_runtime.h>
#include <stdint.h>

typedef _Float16 half_t;
typedef __attribute__((ext_vector_type(16))) _Float16 v16h;
typedef __attribute__((ext_vector_type(8)))  float    v8f;
typedef __attribute__((ext_vector_type(4)))  int      v4i_t;

#define LDA 40          // LDS row stride in halves (80B: 16B-aligned, bank-friendly)
#define BN_EPS 0.001f

// ---- CDNA5 async global->LDS (guarded: falls back to reg staging if absent) ----
#if defined(__has_builtin)
#if __has_builtin(__builtin_amdgcn_global_load_async_to_lds_b128)
#define USE_ASYNC_LDS 1
#endif
#endif

#if defined(USE_ASYNC_LDS)
#define AS1 __attribute__((address_space(1)))
#define AS3 __attribute__((address_space(3)))
#if __has_builtin(__builtin_amdgcn_s_wait_asynccnt)
#define WAIT_ASYNC() __builtin_amdgcn_s_wait_asynccnt(0)
#else
#define WAIT_ASYNC() asm volatile("s_wait_asynccnt 0x0" ::: "memory")
#endif
#else
#define WAIT_ASYNC()
#endif

// ---------------------------------------------------------------------------
// Helper kernels
// ---------------------------------------------------------------------------
__global__ void fill_u32_kernel(uint32_t* p, uint32_t v, long n) {
  long i = (long)blockIdx.x * blockDim.x + threadIdx.x;
  long stride = (long)gridDim.x * blockDim.x;
  for (; i < n; i += stride) p[i] = v;
}

__global__ void scatter_kernel(const float* __restrict__ feats,
                               const int* __restrict__ coords,
                               half_t* __restrict__ dense, float* __restrict__ mask,
                               int nvox, int D, int H, int W) {
  int i = blockIdx.x * blockDim.x + threadIdx.x;
  if (i >= nvox) return;
  int z = coords[i * 4 + 1];
  int y = coords[i * 4 + 2];
  int x = coords[i * 4 + 3];
  int vid = (z * H + y) * W + x;
  for (int c = 0; c < 4; ++c) dense[vid * 4 + c] = (half_t)feats[i * 4 + c];
  mask[vid] = 1.0f;
}

__global__ void bn_prep_kernel(const float* __restrict__ g, const float* __restrict__ b,
                               const float* __restrict__ mn, const float* __restrict__ vr,
                               float* __restrict__ scale, float* __restrict__ bias, int cout) {
  int i = threadIdx.x;
  if (i >= cout) return;
  float inv = g[i] * rsqrtf(vr[i] + BN_EPS);
  scale[i] = inv;
  bias[i]  = b[i] - mn[i] * inv;
}

// Pack f32 weights (K-major x cout) into per-lane WMMA B fragments:
// packed[((kt*nnt+nt)*32+lane)*16 + j] = W[(kt*32+(lane/16)*16+j)*cout + nt*16+lane%16]
__global__ void pack_w_kernel(const float* __restrict__ w, half_t* __restrict__ out,
                              int Ktot, int cout, int nnt, int total) {
  int tid = blockIdx.x * blockDim.x + threadIdx.x;
  if (tid >= total) return;
  int j    = tid & 15;
  int lane = (tid >> 4) & 31;
  int rest = tid >> 9;
  int nt = rest % nnt;
  int kt = rest / nnt;
  int K  = kt * 32 + (lane >> 4) * 16 + j;
  int co = nt * 16 + (lane & 15);
  float v = (K < Ktot && co < cout) ? w[K * cout + co] : 0.0f;
  out[tid] = (half_t)v;
}

// Occupancy mask for strided layers: OR over receptive window.
__global__ void mask_conv_kernel(const float* __restrict__ min_, float* __restrict__ mout,
                                 int ID, int IH, int IW, int OD, int OH, int OW,
                                 int kd, int kh, int kw, int sd, int sh, int sw,
                                 int pd, int ph, int pw, int Mtot) {
  unsigned m = blockIdx.x * blockDim.x + threadIdx.x;
  if (m >= (unsigned)Mtot) return;
  unsigned ow = m % (unsigned)OW;
  unsigned t  = m / (unsigned)OW;
  unsigned oh = t % (unsigned)OH;
  unsigned od = t / (unsigned)OH;
  int id0 = (int)od * sd - pd, ih0 = (int)oh * sh - ph, iw0 = (int)ow * sw - pw;
  float any = 0.0f;
  for (int dd = 0; dd < kd; ++dd) {
    int id = id0 + dd;
    if ((unsigned)id >= (unsigned)ID) continue;
    for (int hh = 0; hh < kh; ++hh) {
      int ih = ih0 + hh;
      if ((unsigned)ih >= (unsigned)IH) continue;
      for (int wwi = 0; wwi < kw; ++wwi) {
        int iw = iw0 + wwi;
        if ((unsigned)iw >= (unsigned)IW) continue;
        if (min_[(id * IH + ih) * IW + iw] > 0.5f) any = 1.0f;
      }
    }
  }
  mout[m] = any;
}

// ---------------------------------------------------------------------------
// Implicit-GEMM conv3d + BN + ReLU + mask, f16 WMMA (16x16x32, f32 accum)
// 256 threads = 8 waves.  cout<=16: 8(M)x1(N) waves, tile 128x16.
//                         else    : 4(M)x2(N) waves, tile  64x32.
// Double-buffered LDS A-tile (im2col) staged with async global->LDS when
// available; packed global B fragments direct to VGPRs.
// ---------------------------------------------------------------------------
struct ConvParams {
  const half_t* __restrict__ in;
  const half_t* __restrict__ wpk;   // packed B fragments
  const float*  __restrict__ scale;
  const float*  __restrict__ bias;
  const float*  __restrict__ mask;  // occupancy at OUTPUT resolution
  half_t* out_h;                    // f16 ping-pong output (or null)
  float*  out_f;                    // f32 final output (or null)
  int ID, IH, IW, OD, OH, OW;
  int cout;
  int cinMask, cinShift;            // cin is a power of two
  int is3x3;                        // 1: 3x3x3 window, 0: 3x1x1 window
  int sd, sh, sw, pd, ph, pw;
  int Ktot, nkt, nnt;
  int nShift;                       // log2(wavesN): 0 or 1
  int spCount;                      // tileM/64: 1 or 2
  int Mtot;
};

__global__ __launch_bounds__(256) void conv_wmma_kernel(ConvParams p) {
  __shared__ __align__(16) half_t As[2][128 * LDA];

  const int tid  = threadIdx.x;
  const int wave = tid >> 5;
  const int lane = tid & 31;
  const int wn   = wave & ((1 << p.nShift) - 1);
  const int wm   = wave >> p.nShift;
  const int tileM = p.spCount << 6;
  const int m0   = blockIdx.x * tileM;
  const int n0blk = blockIdx.y * (16 << p.nShift);

  // --- A-tile loader assignment: thread -> (row, K-octet); spCount row passes ---
  const int arow   = tid >> 2;          // 0..63
  const int achunk = (tid & 3) * 8;     // 0,8,16,24
  int id0A[2], ih0A[2], iw0A[2];
  bool mvA[2];
  for (int s = 0; s < p.spCount; ++s) {
    int m = m0 + (s << 6) + arow;
    bool mv = m < p.Mtot;
    unsigned mm = mv ? (unsigned)m : 0u;
    unsigned ow = mm % (unsigned)p.OW;
    unsigned t  = mm / (unsigned)p.OW;
    unsigned oh = t % (unsigned)p.OH;
    unsigned od = t / (unsigned)p.OH;
    id0A[s] = (int)od * p.sd - p.pd;
    ih0A[s] = (int)oh * p.sh - p.ph;
    iw0A[s] = (int)ow * p.sw - p.pw;
    mvA[s]  = mv;
  }
  const bool fastA = p.cinMask >= 7;     // cin >= 8 (power of two)

  // --- B fragment base: this wave's N-tile of packed weights ---
  const int ntile = (n0blk >> 4) + wn;
  const int ntcl  = (ntile * 16 < p.cout) ? ntile : 0;   // clamp (results discarded)
  const int bstride = p.nnt * 512;                       // halves per K-step
  const half_t* bp = p.wpk + (ntcl * 32 + lane) * 16;

  // stage one 32-wide K slab of the im2col A tile into LDS buffer `buf`
  auto stage = [&](int kt, int buf) {
    const int Kb = kt * 32 + achunk;
    for (int s = 0; s < p.spCount; ++s) {
      half_t* ldst = &As[buf][((s << 6) + arow) * LDA + achunk];
      if (fastA) {
        const half_t* src = nullptr;
        if (mvA[s] && Kb < p.Ktot) {
          int kpos = Kb >> p.cinShift;
          int ci   = Kb & p.cinMask;
          int kdd, khh, kww;
          if (p.is3x3) {               // kpos < 27: /9 and /3 via mul-shift
            kdd = (kpos * 57) >> 9;
            int r = kpos - kdd * 9;
            khh = (r * 11) >> 5;
            kww = r - khh * 3;
          } else { kdd = kpos; khh = 0; kww = 0; }
          int id = id0A[s] + kdd, ih = ih0A[s] + khh, iw = iw0A[s] + kww;
          if ((unsigned)id < (unsigned)p.ID && (unsigned)ih < (unsigned)p.IH &&
              (unsigned)iw < (unsigned)p.IW) {
            src = p.in + ((((id * p.IH + ih) * p.IW + iw) << p.cinShift) + ci);
          }
        }
#if defined(USE_ASYNC_LDS)
        if (src) {
          // async DMA: global b128 -> LDS, tracked on ASYNCcnt
          __builtin_amdgcn_global_load_async_to_lds_b128(
              (AS1 v4i_t*)src, (AS3 v4i_t*)ldst, 0, 0);
        } else {
          *(float4*)ldst = make_float4(0.f, 0.f, 0.f, 0.f);  // zero OOB chunk
        }
#else
        float4 v = make_float4(0.f, 0.f, 0.f, 0.f);
        if (src) v = *(const float4*)src;
        *(float4*)ldst = v;
#endif
      } else {                          // cin == 4 (layer 1 only): scalar gather
        union { float4 f4; half_t h[8]; } abuf;
        abuf.f4 = make_float4(0.f, 0.f, 0.f, 0.f);
        if (mvA[s] && Kb < p.Ktot) {
          for (int j = 0; j < 8; ++j) {
            int Kj = Kb + j;
            if (Kj >= p.Ktot) break;
            int kpos = Kj >> p.cinShift;
            int ci   = Kj & p.cinMask;
            int kdd = (kpos * 57) >> 9;
            int r = kpos - kdd * 9;
            int khh = (r * 11) >> 5;
            int kww = r - khh * 3;
            int id = id0A[s] + kdd, ih = ih0A[s] + khh, iw = iw0A[s] + kww;
            if ((unsigned)id < (unsigned)p.ID && (unsigned)ih < (unsigned)p.IH &&
                (unsigned)iw < (unsigned)p.IW) {
              abuf.h[j] = p.in[(((id * p.IH + ih) * p.IW + iw) << p.cinShift) + ci];
            }
          }
        }
        *(float4*)ldst = abuf.f4;
      }
    }
  };

  v8f acc = {};
  const int mrow = (wm << 4) + (lane & 15);
  const int kloc = (lane >> 4) * 16;

  stage(0, 0);
  WAIT_ASYNC();
  __syncthreads();

#pragma unroll 2
  for (int kt = 0; kt < p.nkt; ++kt) {
    const int cur = kt & 1;

    // ---- B fragment for this K-step: two contiguous b128 loads ----
    union { v16h v; float4 f[2]; } bfrag;
    const half_t* bcur = bp + kt * bstride;
    bfrag.f[0] = *(const float4*)(bcur);
    bfrag.f[1] = *(const float4*)(bcur + 8);
    if (kt + 1 < p.nkt)
      __builtin_prefetch((const void*)(bcur + bstride), 0, 0);

    // ---- overlap: stage next K slab into the other LDS buffer ----
    if (kt + 1 < p.nkt) stage(kt + 1, cur ^ 1);

    // ---- A fragment from LDS (16-bit A 16x32 lane layout) ----
    union { v16h v; float4 f[2]; } afrag;
    const half_t* ap = &As[cur][mrow * LDA + kloc];
    afrag.f[0] = *(const float4*)(ap);
    afrag.f[1] = *(const float4*)(ap + 8);

    acc = __builtin_amdgcn_wmma_f32_16x16x32_f16(
        /*neg_a=*/false, afrag.v, /*neg_b=*/false, bfrag.v,
        /*c_mod=*/(short)0, acc, /*reuse_a=*/false, /*reuse_b=*/false);

    WAIT_ASYNC();
    __syncthreads();
  }

  // ---- epilogue: BN + ReLU + mask ----
  const int co = n0blk + (wn << 4) + (lane & 15);
  if (co < p.cout) {
    const float sc = p.scale[co];
    const float bi = p.bias[co];
    const int mbase = m0 + (wm << 4) + (lane >> 4) * 8;
    if (p.out_f) {
      for (int r = 0; r < 8; ++r) {
        int m = mbase + r;
        if (m < p.Mtot) {
          float v = acc[r] * sc + bi;
          v = v > 0.f ? v : 0.f;
          p.out_f[m * p.cout + co] = v * p.mask[m];
        }
      }
    } else {
      for (int r = 0; r < 8; ++r) {
        int m = mbase + r;
        if (m < p.Mtot) {
          float v = acc[r] * sc + bi;
          v = v > 0.f ? v : 0.f;
          p.out_h[m * p.cout + co] = (half_t)(v * p.mask[m]);
        }
      }
    }
  }
}

// ---------------------------------------------------------------------------
// Host orchestration
// ---------------------------------------------------------------------------
extern "C" void kernel_launch(void* const* d_in, const int* in_sizes, int n_in,
                              void* d_out, int out_size, void* d_ws, size_t ws_size,
                              hipStream_t stream) {
  (void)in_sizes; (void)n_in; (void)out_size; (void)ws_size;

  const float* feats  = (const float*)d_in[0];
  const int*   coords = (const int*)d_in[1];

  struct Cfg { int cin, cout, kd, kh, kw, sd, sh, sw, pd, ph, pw, subm; };
  static const Cfg cfg[12] = {
    { 4, 16, 3,3,3, 1,1,1, 1,1,1, 1},
    {16, 16, 3,3,3, 1,1,1, 1,1,1, 1},
    {16, 32, 3,3,3, 2,2,2, 1,1,1, 0},
    {32, 32, 3,3,3, 1,1,1, 1,1,1, 1},
    {32, 32, 3,3,3, 1,1,1, 1,1,1, 1},
    {32, 64, 3,3,3, 2,2,2, 1,1,1, 0},
    {64, 64, 3,3,3, 1,1,1, 1,1,1, 1},
    {64, 64, 3,3,3, 1,1,1, 1,1,1, 1},
    {64, 64, 3,3,3, 2,2,2, 0,1,1, 0},
    {64, 64, 3,3,3, 1,1,1, 1,1,1, 1},
    {64, 64, 3,3,3, 1,1,1, 1,1,1, 1},
    {64,128, 3,1,1, 2,1,1, 0,0,0, 0},
  };

  const long MAXVOX  = 41L * 400 * 352;   // 5,772,800
  const long ACT_MAX = MAXVOX * 16;       // 92,364,800 halves (layers 1-2 out)

  char* ws = (char*)d_ws;
  size_t off = 0;
  auto carve = [&](size_t bytes) -> void* {
    off = (off + 255) & ~(size_t)255;
    void* p = ws + off;
    off += bytes;
    return p;
  };
  half_t* act0   = (half_t*)carve((size_t)ACT_MAX * 2);
  half_t* act1   = (half_t*)carve((size_t)ACT_MAX * 2);
  float*  mask0  = (float*)carve((size_t)MAXVOX * 4);
  float*  mask1  = (float*)carve((size_t)MAXVOX * 4);
  float*  bnscale= (float*)carve(12 * 128 * 4);
  float*  bnbias = (float*)carve(12 * 128 * 4);

  size_t woff[12]; size_t wtot = 0;
  int nktA[12], nntA[12], KtotA[12];
  for (int L = 0; L < 12; ++L) {
    int Ktot = cfg[L].kd * cfg[L].kh * cfg[L].kw * cfg[L].cin;
    int nkt  = (Ktot + 31) / 32;
    int nnt  = (cfg[L].cout + 15) / 16;
    KtotA[L] = Ktot; nktA[L] = nkt; nntA[L] = nnt;
    woff[L]  = wtot;
    wtot    += (size_t)nkt * nnt * 512;
  }
  half_t* wpk = (half_t*)carve(wtot * 2);

  // ---- init dense input + initial mask ----
  fill_u32_kernel<<<1024, 256, 0, stream>>>((uint32_t*)act0, 0u, MAXVOX * 4 / 2);
  fill_u32_kernel<<<1024, 256, 0, stream>>>((uint32_t*)mask0, 0u, MAXVOX);
  scatter_kernel<<<(40000 + 255) / 256, 256, 0, stream>>>(
      feats, coords, act0, mask0, 40000, 41, 400, 352);

  // ---- per-layer prep: fused BN params + packed weight fragments ----
  for (int L = 0; L < 12; ++L) {
    const float* w  = (const float*)d_in[2 + 5 * L + 0];
    const float* g  = (const float*)d_in[2 + 5 * L + 1];
    const float* b  = (const float*)d_in[2 + 5 * L + 2];
    const float* mn = (const float*)d_in[2 + 5 * L + 3];
    const float* vr = (const float*)d_in[2 + 5 * L + 4];
    bn_prep_kernel<<<1, 128, 0, stream>>>(g, b, mn, vr,
                                          bnscale + L * 128, bnbias + L * 128,
                                          cfg[L].cout);
    int tot = nktA[L] * nntA[L] * 512;
    pack_w_kernel<<<(tot + 255) / 256, 256, 0, stream>>>(
        w, wpk + woff[L], KtotA[L], cfg[L].cout, nntA[L], tot);
  }

  // ---- layer pipeline ----
  int ID = 41, IH = 400, IW = 352;
  half_t* bin  = act0;
  half_t* bout = act1;
  float*  mcur = mask0;
  float*  mnxt = mask1;

  for (int L = 0; L < 12; ++L) {
    const Cfg& c = cfg[L];
    int OD, OH, OW;
    if (c.subm) { OD = ID; OH = IH; OW = IW; }
    else {
      OD = (ID + 2 * c.pd - c.kd) / c.sd + 1;
      OH = (IH + 2 * c.ph - c.kh) / c.sh + 1;
      OW = (IW + 2 * c.pw - c.kw) / c.sw + 1;
    }
    int Mtot = OD * OH * OW;

    const float* omask = mcur;
    if (!c.subm) {
      mask_conv_kernel<<<(Mtot + 255) / 256, 256, 0, stream>>>(
          mcur, mnxt, ID, IH, IW, OD, OH, OW,
          c.kd, c.kh, c.kw, c.sd, c.sh, c.sw, c.pd, c.ph, c.pw, Mtot);
      omask = mnxt;
    }

    const int wavesN = (c.cout <= 16) ? 1 : 2;    // 8x1 or 4x2 wave layout
    const int tileM  = (wavesN == 1) ? 128 : 64;

    ConvParams p{};
    p.in = bin; p.wpk = wpk + woff[L];
    p.scale = bnscale + L * 128; p.bias = bnbias + L * 128;
    p.mask = omask;
    const bool last = (L == 11);
    p.out_h = last ? nullptr : bout;
    p.out_f = last ? (float*)d_out : nullptr;
    p.ID = ID; p.IH = IH; p.IW = IW; p.OD = OD; p.OH = OH; p.OW = OW;
    p.cout = c.cout;
    p.cinMask  = c.cin - 1;
    p.cinShift = __builtin_ctz((unsigned)c.cin);
    p.is3x3 = (c.kh == 3) ? 1 : 0;
    p.sd = c.sd; p.sh = c.sh; p.sw = c.sw;
    p.pd = c.pd; p.ph = c.ph; p.pw = c.pw;
    p.Ktot = KtotA[L]; p.nkt = nktA[L]; p.nnt = nntA[L];
    p.nShift = (wavesN == 2) ? 1 : 0;
    p.spCount = tileM / 64;
    p.Mtot = Mtot;

    dim3 grid((unsigned)((Mtot + tileM - 1) / tileM),
              (unsigned)((c.cout + 16 * wavesN - 1) / (16 * wavesN)));
    conv_wmma_kernel<<<grid, 256, 0, stream>>>(p);

    { half_t* t = bin; bin = bout; bout = t; }
    if (!c.subm) { float* t = mcur; mcur = mnxt; mnxt = t; }
    ID = OD; IH = OH; IW = OW;
  }
}
